// MultiHeadAttention_48043504173545
// MI455X (gfx1250) — matrix operand order repaired and checked by reading, not compiled, not run on hardware
//
#include <hip/hip_runtime.h>

typedef __attribute__((ext_vector_type(16))) _Float16 v16h;
typedef __attribute__((ext_vector_type(8)))  float    v8f;
typedef __attribute__((ext_vector_type(4)))  unsigned int v4u;
typedef __attribute__((ext_vector_type(4)))  int      v4i;
typedef __attribute__((ext_vector_type(8)))  int      v8i;

#define EMBED    1024
#define HEADS    16
#define HEAD_DIM 64
#define BATCH    4
#define SEQ      2048
#define MTOT     (BATCH * SEQ)   /* 8192 token rows */

union FragU {
    v16h v;
    unsigned int u[8];
    _Float16 h[16];
};

static __device__ inline v8f wmma16(v16h a, v16h b, v8f c) {
    // D = A(16x32 f16) * B(32x16 f16) + C(16x16 f32)
    return __builtin_amdgcn_wmma_f32_16x16x32_f16(false, a, false, b,
                                                  (short)0, c, false, false);
}

// ---------------------------------------------------------------------------
// TDM: DMA a 2D f16 tile (tile_d0 contiguous elems x tile_d1 rows) from global
// to LDS, with hardware padding: 16 DWORDs (64B = one 32-half tile row) then
// 4 DWORDs pad -> effective LDS row stride of 40 halves (conflict-free frags).
// D# packing per CDNA5 ISA ch. 8 (group0: count/lds/global/type, group1:
// data_size, pad ctl, tensor dims, tile dims, dim0 stride).
// This toolchain exposes the 6-arg builtin:
//   (uint32x4 g0, int32x8 g1, int32x4 g2, int32x4 g3, int32x8 extra, i32 cpol)
// ---------------------------------------------------------------------------
static __device__ inline void tdm_load_tile_f16(unsigned int lds_off,
                                                const _Float16* gptr,
                                                unsigned int tensor_d0,
                                                unsigned int tensor_d1,
                                                unsigned int tile_d0,
                                                unsigned int tile_d1,
                                                unsigned int stride_elems)
{
    const unsigned long long ga = (unsigned long long)(uintptr_t)gptr;
    v4u g0;
    g0[0] = 1u;                                        // count=1, user mode
    g0[1] = lds_off;                                   // lds_addr (bytes)
    g0[2] = (unsigned int)(ga & 0xFFFFFFFFull);        // global_addr[31:0]
    g0[3] = (unsigned int)((ga >> 32) & 0x01FFFFFFull) // global_addr[56:32]
          | (2u << 30);                                // type = 2 ("image")

    const unsigned int w0 = (1u << 16)   // data_size = 1 -> 2 bytes
                          | (1u << 20)   // pad_enable
                          | (3u << 22)   // pad_interval code 3 = 16 DWORDs
                          | (3u << 25);  // pad_amount   code 3 = 4 DWORDs
    v8i g1;
    g1[0] = (int)w0;
    g1[1] = (int)(tensor_d0 << 16);                       // td0[15:0]
    g1[2] = (int)((tensor_d0 >> 16) | (tensor_d1 << 16)); // td0[31:16], td1[15:0]
    g1[3] = (int)((tensor_d1 >> 16) | (tile_d0 << 16));   // td1[31:16], tile0
    g1[4] = (int)tile_d1;                                 // tile1; tile2 = 0
    g1[5] = (int)stride_elems;                            // dim0_stride[31:0]
    g1[6] = 0;
    g1[7] = 0;
    const v4i z4 = {};
    const v8i z8 = {};
    __builtin_amdgcn_tensor_load_to_lds(g0, g1, z4, z4, z8, 0);
}

// ---------------------------------------------------------------------------
// f32 -> f16 elementwise convert (8 elems/thread, b128 in / b128 out)
// ---------------------------------------------------------------------------
__global__ __launch_bounds__(256)
void cvt_f16_kernel(const float* __restrict__ in, _Float16* __restrict__ out)
{
    const size_t i = ((size_t)blockIdx.x * 256 + threadIdx.x) * 8;
    const float4 a = *(const float4*)(in + i);
    const float4 b = *(const float4*)(in + i + 4);
    union { _Float16 h[8]; uint4 u; } p;
    p.h[0] = (_Float16)a.x; p.h[1] = (_Float16)a.y;
    p.h[2] = (_Float16)a.z; p.h[3] = (_Float16)a.w;
    p.h[4] = (_Float16)b.x; p.h[5] = (_Float16)b.y;
    p.h[6] = (_Float16)b.z; p.h[7] = (_Float16)b.w;
    *(uint4*)(out + i) = p.u;
}

// ---------------------------------------------------------------------------
// GEMM: Out[M=8192, N=1024] = A[M,K] * W[N,K]^T + bias[N]; A,W are f16.
// Block tile 128x64, 8 waves of 32x32, K-step 32.
// Staging: double-buffered TDM (wave 0 issues 2 descriptors per step,
// s_wait_tensorcnt pipelines DMA against WMMA).
// OMODE: 0 = f32 row-major, 1 = f16 row-major, 2 = f16 per-head transposed
//        ([b][h][d][s] — used for V so attention B-frags are contiguous)
// ---------------------------------------------------------------------------
template<int OMODE>
__global__ __launch_bounds__(256)
void gemm_wmma_kernel(const _Float16* __restrict__ Ah,
                      const _Float16* __restrict__ Wh,
                      const float* __restrict__ bias,
                      void* __restrict__ Out)
{
    constexpr int K   = EMBED;
    constexpr int N   = EMBED;
    constexpr int LDT = 40;            // LDS row stride in halves (TDM padded)

    __shared__ _Float16 lA[2][128 * LDT];
    __shared__ _Float16 lB[2][64 * LDT];

    const int tid  = threadIdx.x;
    const int wave = tid >> 5;
    const int lane = tid & 31;
    const int ln   = lane & 15;
    const int hi   = lane >> 4;
    const int m0   = blockIdx.x * 128;
    const int n0   = blockIdx.y * 64;
    const int wm   = (wave >> 1) * 32;
    const int wn   = (wave & 1) * 32;

    v8f acc[2][2];
    const v8f zero = {};
    acc[0][0] = zero; acc[0][1] = zero; acc[1][0] = zero; acc[1][1] = zero;

    // issue both tiles of one k-step into buffer `buf`
    auto issue = [&](int buf, int k0) {
        tdm_load_tile_f16((unsigned int)(uintptr_t)&lA[buf][0],
                          Ah + (size_t)m0 * K + k0, K, MTOT, 32, 128, K);
        tdm_load_tile_f16((unsigned int)(uintptr_t)&lB[buf][0],
                          Wh + (size_t)n0 * K + k0, K, N, 32, 64, K);
    };

    if (wave == 0) issue(0, 0);

    constexpr int NSTEP = K / 32;
    for (int i = 0; i < NSTEP; ++i) {
        if (wave == 0) {
            if (i + 1 < NSTEP) {
                issue((i + 1) & 1, (i + 1) * 32);
                __builtin_amdgcn_s_wait_tensorcnt(2);  // buffer i complete
            } else {
                __builtin_amdgcn_s_wait_tensorcnt(0);
            }
        }
        __syncthreads();

        const _Float16* la = &lA[i & 1][0];
        const _Float16* lb = &lB[i & 1][0];

        FragU af[2], bf[2];
#pragma unroll
        for (int t = 0; t < 2; ++t) {
            const int row = wm + t * 16 + ln;
#pragma unroll
            for (int j = 0; j < 8; ++j) {
                const int kk = 2 * j + hi * 16;
                af[t].u[j] = *(const unsigned int*)&la[row * LDT + kk];
            }
        }
#pragma unroll
        for (int t = 0; t < 2; ++t) {
            const int row = wn + t * 16 + ln;
#pragma unroll
            for (int j = 0; j < 8; ++j) {
                const int kk = 2 * j + hi * 16;
                bf[t].u[j] = *(const unsigned int*)&lb[row * LDT + kk];
            }
        }
#pragma unroll
        for (int t = 0; t < 2; ++t)
#pragma unroll
            for (int j = 0; j < 2; ++j)
                acc[t][j] = wmma16(af[t].v, bf[j].v, acc[t][j]);

        __syncthreads();
    }

    // ---- epilogue: bias add + store
    float bv[2];
#pragma unroll
    for (int j = 0; j < 2; ++j) bv[j] = bias[n0 + wn + j * 16 + ln];

#pragma unroll
    for (int t = 0; t < 2; ++t)
#pragma unroll
        for (int j = 0; j < 2; ++j)
#pragma unroll
            for (int r = 0; r < 8; ++r) {
                const int gm = m0 + wm + t * 16 + r + hi * 8;
                const int gn = n0 + wn + j * 16 + ln;
                const float val = acc[t][j][r] + bv[j];
                if (OMODE == 0) {
                    ((float*)Out)[(size_t)gm * N + gn] = val;
                } else if (OMODE == 1) {
                    ((_Float16*)Out)[(size_t)gm * N + gn] = (_Float16)val;
                } else {
                    const int bb = gm / SEQ, s = gm % SEQ;
                    const int hh = gn >> 6,  d = gn & 63;
                    ((_Float16*)Out)[(((size_t)bb * HEADS + hh) * HEAD_DIM + d) * SEQ + s] =
                        (_Float16)val;
                }
            }
}

// ---------------------------------------------------------------------------
// Flash attention: one wave owns one 16-query block of one (b,h).
// Qp, Kp: f16 [b][s][h*64+d]; Vt: f16 [b][h][d][s]; Op: f16 [b][s][h*64+d].
// Scale 1/sqrt(64) folded into Q fragments (exact: power of two).
// ---------------------------------------------------------------------------
__global__ __launch_bounds__(256)
void flash_attn_kernel(const _Float16* __restrict__ Qp,
                       const _Float16* __restrict__ Kp,
                       const _Float16* __restrict__ Vt,
                       _Float16* __restrict__ Op)
{
    __shared__ _Float16 pbuf[8 * 16 * 34];   // per-wave P transpose buffer

    const int tid  = threadIdx.x;
    const int wave = tid >> 5;
    const int lane = tid & 31;
    const int ln   = lane & 15;
    const int hi   = lane >> 4;

    const int qb = blockIdx.x * 8 + wave;        // global 16-query block id
    const int q0 = (qb & (SEQ / 16 - 1)) * 16;
    const int h  = (qb >> 7) & (HEADS - 1);
    const int b  = qb >> 11;

    const size_t bh = (size_t)b * SEQ * EMBED + (size_t)h * HEAD_DIM;
    const _Float16* vbase = Vt + ((size_t)b * HEADS + h) * HEAD_DIM * SEQ;
    _Float16* pb = &pbuf[wave * 16 * 34];

    // ---- preload Q fragments (16x64 = two 16x32 A tiles), scale folded in
    FragU qa[2];
    {
        const _Float16* qptr = Qp + bh + (size_t)(q0 + ln) * EMBED;
#pragma unroll
        for (int f = 0; f < 2; ++f) {
#pragma unroll
            for (int j = 0; j < 8; ++j) {
                const int d = f * 32 + 2 * j + hi * 16;
                qa[f].u[j] = *(const unsigned int*)(qptr + d);
            }
            qa[f].v *= (_Float16)0.125f;   // 1/sqrt(HEAD_DIM)
        }
    }

    float mrow[8], lrow[8];
    v8f o[4];
    const v8f zero = {};
#pragma unroll
    for (int r = 0; r < 8; ++r) { mrow[r] = -1e30f; lrow[r] = 0.0f; }
#pragma unroll
    for (int d = 0; d < 4; ++d) o[d] = zero;

    for (int kt = 0; kt < SEQ; kt += 32) {
        // ---- S = (Q*scale) K^T for 32 keys (two 16-key tiles), f32 accum
        v8f s0 = zero, s1 = zero;
        {
            FragU kb;
            const _Float16* kp0 = Kp + bh + (size_t)(kt + ln) * EMBED + hi * 16;
            kb.v = *(const v16h*)kp0;        s0 = wmma16(qa[0].v, kb.v, s0);
            kb.v = *(const v16h*)(kp0 + 32); s0 = wmma16(qa[1].v, kb.v, s0);
            const _Float16* kp1 = kp0 + 16 * EMBED;
            kb.v = *(const v16h*)kp1;        s1 = wmma16(qa[0].v, kb.v, s1);
            kb.v = *(const v16h*)(kp1 + 32); s1 = wmma16(qa[1].v, kb.v, s1);
        }

        // ---- online-softmax row stats (rows live in one 16-lane half)
        float nm[8], rs[8];
#pragma unroll
        for (int r = 0; r < 8; ++r)
            nm[r] = fmaxf(mrow[r], fmaxf(s0[r], s1[r]));
#pragma unroll
        for (int mask = 1; mask <= 8; mask <<= 1)
#pragma unroll
            for (int r = 0; r < 8; ++r)
                nm[r] = fmaxf(nm[r], __shfl_xor(nm[r], mask, 32));
#pragma unroll
        for (int r = 0; r < 8; ++r) {
            const float corr = __expf(mrow[r] - nm[r]);
            s0[r] = __expf(s0[r] - nm[r]);
            s1[r] = __expf(s1[r] - nm[r]);
            rs[r] = s0[r] + s1[r];
            lrow[r] *= corr;
            mrow[r] = nm[r];
#pragma unroll
            for (int d = 0; d < 4; ++d) o[d][r] *= corr;
        }
#pragma unroll
        for (int mask = 1; mask <= 8; mask <<= 1)
#pragma unroll
            for (int r = 0; r < 8; ++r)
                rs[r] += __shfl_xor(rs[r], mask, 32);
#pragma unroll
        for (int r = 0; r < 8; ++r) lrow[r] += rs[r];

        // ---- P: C-layout -> A-layout via per-wave LDS (in-order per wave)
#pragma unroll
        for (int r = 0; r < 8; ++r) {
            pb[(r + hi * 8) * 34 + ln]      = (_Float16)s0[r];
            pb[(r + hi * 8) * 34 + 16 + ln] = (_Float16)s1[r];
        }
        FragU pa;
#pragma unroll
        for (int j = 0; j < 8; ++j) {
            const int kk = 2 * j + hi * 16;
            pa.u[j] = *(const unsigned int*)&pb[ln * 34 + kk];
        }

        // ---- O += P @ V (V transposed: key dim contiguous)
#pragma unroll
        for (int d = 0; d < 4; ++d) {
            FragU vb;
            const _Float16* vcol = vbase + (size_t)(d * 16 + ln) * SEQ + kt + hi * 16;
#pragma unroll
            for (int j = 0; j < 8; ++j)
                vb.u[j] = *(const unsigned int*)(vcol + 2 * j);
            o[d] = wmma16(pa.v, vb.v, o[d]);
        }
    }

    // ---- normalize and store (f16, [b][s][h*64+d] so final GEMM reads it)
#pragma unroll
    for (int r = 0; r < 8; ++r) {
        const float inv = 1.0f / lrow[r];
        const int q = q0 + r + hi * 8;
#pragma unroll
        for (int d = 0; d < 4; ++d)
            Op[(size_t)(b * SEQ + q) * EMBED + h * HEAD_DIM + d * 16 + ln] =
                (_Float16)(o[d][r] * inv);
    }
}

// ---------------------------------------------------------------------------
extern "C" void kernel_launch(void* const* d_in, const int* in_sizes, int n_in,
                              void* d_out, int out_size, void* d_ws, size_t ws_size,
                              hipStream_t stream)
{
    (void)in_sizes; (void)n_in; (void)out_size; (void)ws_size;
    const float* q  = (const float*)d_in[0];
    const float* k  = (const float*)d_in[1];
    const float* v  = (const float*)d_in[2];
    const float* Wq = (const float*)d_in[3];
    const float* bq = (const float*)d_in[4];
    const float* Wk = (const float*)d_in[5];
    const float* bk = (const float*)d_in[6];
    const float* Wv = (const float*)d_in[7];
    const float* bv = (const float*)d_in[8];
    const float* Wo = (const float*)d_in[9];
    const float* bo = (const float*)d_in[10];

    const size_t ael = (size_t)MTOT * EMBED;    // activation elems
    const size_t wel = (size_t)EMBED * EMBED;   // weight elems

    _Float16* Xq  = (_Float16*)d_ws;            // f16 copies of inputs
    _Float16* Xk  = Xq  + ael;
    _Float16* Xv  = Xk  + ael;
    _Float16* Wqh = Xv  + ael;                  // f16 copies of weights
    _Float16* Wkh = Wqh + wel;
    _Float16* Wvh = Wkh + wel;
    _Float16* Woh = Wvh + wel;
    _Float16* Qp  = Woh + wel;                  // f16 [b][s][e]
    _Float16* Kp  = Qp  + ael;                  // f16 [b][s][e]
    _Float16* Vt  = Kp  + ael;                  // f16 [b][h][d][s]
    _Float16* AO  = Vt  + ael;                  // f16 [b][s][e]

    // f32 -> f16 pre-conversion (8 elems per thread)
    cvt_f16_kernel<<<ael / 2048, 256, 0, stream>>>(q,  Xq);
    cvt_f16_kernel<<<ael / 2048, 256, 0, stream>>>(k,  Xk);
    cvt_f16_kernel<<<ael / 2048, 256, 0, stream>>>(v,  Xv);
    cvt_f16_kernel<<<wel / 2048, 256, 0, stream>>>(Wq, Wqh);
    cvt_f16_kernel<<<wel / 2048, 256, 0, stream>>>(Wk, Wkh);
    cvt_f16_kernel<<<wel / 2048, 256, 0, stream>>>(Wv, Wvh);
    cvt_f16_kernel<<<wel / 2048, 256, 0, stream>>>(Wo, Woh);

    dim3 grid(MTOT / 128, EMBED / 64);
    gemm_wmma_kernel<1><<<grid, 256, 0, stream>>>(Xq, Wqh, bq, Qp);
    gemm_wmma_kernel<1><<<grid, 256, 0, stream>>>(Xk, Wkh, bk, Kp);
    gemm_wmma_kernel<2><<<grid, 256, 0, stream>>>(Xv, Wvh, bv, Vt);

    flash_attn_kernel<<<(BATCH * HEADS * (SEQ / 16)) / 8, 256, 0, stream>>>(Qp, Kp, Vt, AO);

    gemm_wmma_kernel<0><<<grid, 256, 0, stream>>>(AO, Woh, bo, d_out);
}
